// Transformer_63239098466924
// MI455X (gfx1250) — matrix operand order repaired and checked by reading, not compiled, not run on hardware
//
#include <hip/hip_runtime.h>
#include <hip/hip_bf16.h>
#include <float.h>
#include <math.h>

// ---------------------------------------------------------------------------
// GraphTransformer (2x PyG TransformerConv) for MI455X (gfx1250, wave32).
// GEMMs run on v_wmma_f32_16x16x32_bf16; edge softmax via float atomics.
// ---------------------------------------------------------------------------

#define NN 50000
#define NE 800000
#define D_IN 512
#define D_MID 64   // 8 heads * 8 ch
#define D_OUT 64

typedef __attribute__((ext_vector_type(16))) __bf16 v16bf;
typedef __attribute__((ext_vector_type(8)))  float  v8f;

// ---------------- weight prep: f32 (K x 64) -> bf16 transposed (64 x K) ----
__global__ void wprep_kernel(const float* __restrict__ W, __bf16* __restrict__ Wt, int K) {
    int total = K * 64;
    for (int t = blockIdx.x * blockDim.x + threadIdx.x; t < total; t += gridDim.x * blockDim.x) {
        int n = t / K;          // output column (row of Wt)
        int k = t - n * K;      // input feature
        Wt[t] = (__bf16)W[(size_t)k * 64 + n];
    }
}

__global__ void fillf_kernel(float* __restrict__ p, float v, int n) {
    for (int i = blockIdx.x * blockDim.x + threadIdx.x; i < n; i += gridDim.x * blockDim.x)
        p[i] = v;
}

// ---------------- fused 4-matrix GEMM:  X(N x K) @ W*(K x 64) + b* ----------
// One block = 512 threads = 16 waves. Block covers 16 rows of X.
// Wave w: matrix (w>>2) in {Q,K,V,Skip}, column tile (w&3) -> 16 cols.
// All 16 waves share one LDS tile of X (bf16), converted on the fly.
template<int K>
__global__ __launch_bounds__(512)
void gemm4_wmma(const float* __restrict__ X, int nrows,
                const __bf16* __restrict__ Wt0, const __bf16* __restrict__ Wt1,
                const __bf16* __restrict__ Wt2, const __bf16* __restrict__ Wt3,
                const float* __restrict__ b0, const float* __restrict__ b1,
                const float* __restrict__ b2, const float* __restrict__ b3,
                float* __restrict__ o0, float* __restrict__ o1,
                float* __restrict__ o2, float* __restrict__ o3)
{
    __shared__ __bf16 lds[16 * K];
    const int rowbase = blockIdx.x * 16;

    // stage 16 x K activation tile into LDS as bf16 (zero-pad tail rows)
    for (int i = threadIdx.x; i < 16 * K; i += 512) {
        int r = i / K, c = i - r * K;
        int gr = rowbase + r;
        float v = (gr < nrows) ? X[(size_t)gr * K + c] : 0.0f;
        lds[i] = (__bf16)v;
    }
    __syncthreads();

    const int wave = threadIdx.x >> 5;
    const int lane = threadIdx.x & 31;
    const int mat  = wave >> 2;
    const int ct   = wave & 3;
    const int colbase = ct * 16;

    const __bf16* Wt = (mat == 0) ? Wt0 : (mat == 1) ? Wt1 : (mat == 2) ? Wt2 : Wt3;
    const float*  bb = (mat == 0) ? b0  : (mat == 1) ? b1  : (mat == 2) ? b2  : b3;
    float*        oo = (mat == 0) ? o0  : (mat == 1) ? o1  : (mat == 2) ? o2  : o3;

    const int half = lane >> 4;    // lane group (ISA 16-bit A/B layout)
    const int mrow = lane & 15;    // A-matrix row for this lane
    const int ncol = lane & 15;    // B/C column for this lane

    // C layout: VGPR i, lanes<16 -> M=i ; lanes>=16 -> M=i+8 ; N = lane&15
    v8f acc;
    const float bias = bb[colbase + ncol];
    #pragma unroll
    for (int i = 0; i < 8; ++i) acc[i] = bias;

    const __bf16* wrow = &Wt[(size_t)(colbase + ncol) * K];

    for (int kc = 0; kc < K; kc += 32) {
        // A frag (16x32 bf16): elems 0..7 = K = kc+half*8+j, elems 8..15 = +16
        union { v16bf v; uint4 u[2]; } au;
        au.u[0] = *(const uint4*)&lds[mrow * K + kc + half * 8];
        au.u[1] = *(const uint4*)&lds[mrow * K + kc + 16 + half * 8];
        // B frag (32x16 bf16): elems 0..7 = K = kc+half*8+j, elems 8..15 = +16
        union { v16bf v; uint4 u[2]; } bu;
        bu.u[0] = *(const uint4*)&wrow[kc + half * 8];
        bu.u[1] = *(const uint4*)&wrow[kc + 16 + half * 8];
        acc = __builtin_amdgcn_wmma_f32_16x16x32_bf16(
                  false, au.v, false, bu.v, (short)0, acc, false, false);
    }

    #pragma unroll
    for (int i = 0; i < 8; ++i) {
        int gr = rowbase + i + 8 * half;
        if (gr < nrows) oo[(size_t)gr * 64 + colbase + ncol] = acc[i];
    }
}

// ---------------- edge phase ------------------------------------------------
__device__ __forceinline__ float atomicMaxF(float* addr, float val) {
    if (val >= 0.0f)
        return __int_as_float(atomicMax((int*)addr, __float_as_int(val)));
    else
        return __uint_as_float(atomicMin((unsigned int*)addr, __float_as_uint(val)));
}

template<int H, int C>
__global__ void edge_logits_kernel(const float* __restrict__ q, const float* __restrict__ k,
                                   const long long* __restrict__ ei,
                                   float* __restrict__ alpha, float* __restrict__ amax)
{
    const float scale = rsqrtf((float)C);
    for (int e = blockIdx.x * blockDim.x + threadIdx.x; e < NE; e += gridDim.x * blockDim.x) {
        const int src = (int)ei[e];
        const int dst = (int)ei[NE + e];
        const float4* qd = (const float4*)&q[(size_t)dst * (H * C)];
        const float4* ks = (const float4*)&k[(size_t)src * (H * C)];
        #pragma unroll
        for (int h = 0; h < H; ++h) {
            float dot = 0.0f;
            #pragma unroll
            for (int c4 = 0; c4 < C / 4; ++c4) {
                float4 a = qd[h * (C / 4) + c4];
                float4 b = ks[h * (C / 4) + c4];
                dot += a.x * b.x + a.y * b.y + a.z * b.z + a.w * b.w;
            }
            float v = dot * scale;
            alpha[(size_t)e * H + h] = v;
            atomicMaxF(&amax[(size_t)dst * H + h], v);
        }
    }
}

template<int H>
__global__ void edge_exp_kernel(const long long* __restrict__ ei,
                                float* __restrict__ alpha,          // in: logits, out: exp
                                const float* __restrict__ amax,
                                float* __restrict__ denom)
{
    for (int e = blockIdx.x * blockDim.x + threadIdx.x; e < NE; e += gridDim.x * blockDim.x) {
        const int dst = (int)ei[NE + e];
        #pragma unroll
        for (int h = 0; h < H; ++h) {
            float ex = expf(alpha[(size_t)e * H + h] - amax[(size_t)dst * H + h]);
            alpha[(size_t)e * H + h] = ex;
            atomicAdd(&denom[(size_t)dst * H + h], ex);
        }
    }
}

template<int H, int C>
__global__ void edge_scatter_kernel(const long long* __restrict__ ei,
                                    const float* __restrict__ ex,
                                    const float* __restrict__ denom,
                                    const float* __restrict__ v,
                                    float* __restrict__ out)   // pre-initialized with skip
{
    for (int e = blockIdx.x * blockDim.x + threadIdx.x; e < NE; e += gridDim.x * blockDim.x) {
        const int src = (int)ei[e];
        const int dst = (int)ei[NE + e];
        #pragma unroll
        for (int h = 0; h < H; ++h) {
            float a = ex[(size_t)e * H + h] / (denom[(size_t)dst * H + h] + 1e-16f);
            #pragma unroll
            for (int c = 0; c < C; ++c) {
                atomicAdd(&out[(size_t)dst * 64 + h * C + c],
                          a * v[(size_t)src * 64 + h * C + c]);
            }
        }
    }
}

// ---------------- elementwise epilogues ------------------------------------
__global__ void elu_kernel(float* __restrict__ p, int n) {
    for (int i = blockIdx.x * blockDim.x + threadIdx.x; i < n; i += gridDim.x * blockDim.x) {
        float x = p[i];
        p[i] = (x > 0.0f) ? x : expm1f(x);
    }
}

// one wave per row of 64 classes; wave32 shuffle reductions
__global__ __launch_bounds__(256)
void logsoftmax_kernel(const float* __restrict__ h, float* __restrict__ out)
{
    const int wave = threadIdx.x >> 5;
    const int lane = threadIdx.x & 31;
    const int row  = blockIdx.x * 8 + wave;
    if (row >= NN) return;
    const float* r = &h[(size_t)row * 64];
    float a = r[lane];
    float b = r[lane + 32];
    float m = fmaxf(a, b);
    #pragma unroll
    for (int s = 16; s > 0; s >>= 1) m = fmaxf(m, __shfl_xor(m, s, 32));
    float se = expf(a - m) + expf(b - m);
    #pragma unroll
    for (int s = 16; s > 0; s >>= 1) se += __shfl_xor(se, s, 32);
    float lse = m + logf(se);
    out[(size_t)row * 64 + lane]      = a - lse;
    out[(size_t)row * 64 + lane + 32] = b - lse;
}

// ---------------------------------------------------------------------------
extern "C" void kernel_launch(void* const* d_in, const int* in_sizes, int n_in,
                              void* d_out, int out_size, void* d_ws, size_t ws_size,
                              hipStream_t stream)
{
    (void)in_sizes; (void)n_in; (void)out_size; (void)ws_size;

    const float*     x  = (const float*)d_in[0];
    const long long* ei = (const long long*)d_in[1];
    const float* W1[4] = { (const float*)d_in[2], (const float*)d_in[4],
                           (const float*)d_in[6], (const float*)d_in[8]  };
    const float* b1[4] = { (const float*)d_in[3], (const float*)d_in[5],
                           (const float*)d_in[7], (const float*)d_in[9]  };
    const float* W2[4] = { (const float*)d_in[10], (const float*)d_in[12],
                           (const float*)d_in[14], (const float*)d_in[16] };
    const float* b2[4] = { (const float*)d_in[11], (const float*)d_in[13],
                           (const float*)d_in[15], (const float*)d_in[17] };

    // ---- scratch carve-out ----
    char* ws = (char*)d_ws;
    size_t off = 0;
    auto carve = [&](size_t bytes) -> void* {
        void* p = ws + off;
        off += (bytes + 255) & ~(size_t)255;
        return p;
    };
    float* q1     = (float*)carve((size_t)NN * 64 * 4);
    float* k1     = (float*)carve((size_t)NN * 64 * 4);
    float* v1     = (float*)carve((size_t)NN * 64 * 4);
    float* h1     = (float*)carve((size_t)NN * 64 * 4);   // skip + attn, then ELU
    float* ex1    = (float*)carve((size_t)NE * 8 * 4);    // logits -> exp (in place)
    float* amax1  = (float*)carve((size_t)NN * 8 * 4);
    float* den1   = (float*)carve((size_t)NN * 8 * 4);
    float* q2     = (float*)carve((size_t)NN * 64 * 4);
    float* k2     = (float*)carve((size_t)NN * 64 * 4);
    float* v2     = (float*)carve((size_t)NN * 64 * 4);
    float* h2     = (float*)carve((size_t)NN * 64 * 4);
    float* ex2    = (float*)carve((size_t)NE * 4);
    float* amax2  = (float*)carve((size_t)NN * 4);
    float* den2   = (float*)carve((size_t)NN * 4);
    __bf16* wt1   = (__bf16*)carve((size_t)4 * D_IN * 64 * 2);  // 4x (64 x 512) bf16
    __bf16* wt2   = (__bf16*)carve((size_t)4 * D_MID * 64 * 2); // 4x (64 x 64)  bf16

    float* outp = (float*)d_out;

    const int TB = 256;
    const int eg = (NE + TB - 1) / TB;

    // ---- weight prep (bf16 transposed) ----
    for (int m = 0; m < 4; ++m) {
        wprep_kernel<<<(D_IN * 64 + TB - 1) / TB, TB, 0, stream>>>(
            W1[m], wt1 + (size_t)m * D_IN * 64, D_IN);
        wprep_kernel<<<(D_MID * 64 + TB - 1) / TB, TB, 0, stream>>>(
            W2[m], wt2 + (size_t)m * D_MID * 64, D_MID);
    }

    // ---- per-call scratch init (deterministic) ----
    fillf_kernel<<<(NN * 8 + TB - 1) / TB, TB, 0, stream>>>(amax1, -FLT_MAX, NN * 8);
    fillf_kernel<<<(NN * 8 + TB - 1) / TB, TB, 0, stream>>>(den1, 0.0f, NN * 8);
    fillf_kernel<<<(NN + TB - 1) / TB, TB, 0, stream>>>(amax2, -FLT_MAX, NN);
    fillf_kernel<<<(NN + TB - 1) / TB, TB, 0, stream>>>(den2, 0.0f, NN);

    const int rowTiles = (NN + 15) / 16;   // 3125

    // ---- layer 1: Q/K/V/skip GEMMs (skip lands in h1) ----
    gemm4_wmma<D_IN><<<rowTiles, 512, 0, stream>>>(
        x, NN,
        wt1 + 0 * (size_t)D_IN * 64, wt1 + 1 * (size_t)D_IN * 64,
        wt1 + 2 * (size_t)D_IN * 64, wt1 + 3 * (size_t)D_IN * 64,
        b1[0], b1[1], b1[2], b1[3],
        q1, k1, v1, h1);

    // ---- layer 1 edge softmax + aggregate (H=8, C=8) ----
    edge_logits_kernel<8, 8><<<eg, TB, 0, stream>>>(q1, k1, ei, ex1, amax1);
    edge_exp_kernel<8><<<eg, TB, 0, stream>>>(ei, ex1, amax1, den1);
    edge_scatter_kernel<8, 8><<<eg, TB, 0, stream>>>(ei, ex1, den1, v1, h1);

    // ---- ELU ----
    elu_kernel<<<(NN * 64 + TB - 1) / TB, TB, 0, stream>>>(h1, NN * 64);

    // ---- layer 2: Q/K/V/skip GEMMs from h1 (skip lands in h2) ----
    gemm4_wmma<D_MID><<<rowTiles, 512, 0, stream>>>(
        h1, NN,
        wt2 + 0 * (size_t)D_MID * 64, wt2 + 1 * (size_t)D_MID * 64,
        wt2 + 2 * (size_t)D_MID * 64, wt2 + 3 * (size_t)D_MID * 64,
        b2[0], b2[1], b2[2], b2[3],
        q2, k2, v2, h2);

    // ---- layer 2 edge softmax + aggregate (H=1, C=64) ----
    edge_logits_kernel<1, 64><<<eg, TB, 0, stream>>>(q2, k2, ei, ex2, amax2);
    edge_exp_kernel<1><<<eg, TB, 0, stream>>>(ei, ex2, amax2, den2);
    edge_scatter_kernel<1, 64><<<eg, TB, 0, stream>>>(ei, ex2, den2, v2, h2);

    // ---- log_softmax over 64 classes -> d_out ----
    logsoftmax_kernel<<<NN / 8, 256, 0, stream>>>(h2, outp);
}
